// UnifiedDiffusionRefiner_52158082842746
// MI455X (gfx1250) — matrix operand order repaired and checked by reading, not compiled, not run on hardware
//
#include <hip/hip_runtime.h>
#include <hip/hip_bf16.h>

typedef __attribute__((ext_vector_type(16))) _Float16 v16h;
typedef __attribute__((ext_vector_type(8)))  _Float16 v8h;
typedef __attribute__((ext_vector_type(8)))  float    v8f;

#define Hdim 96
#define Tdim 32
#define RBFdim 16
#define NLAYERS 4
#define Nn 8192
#define Ee 262144
#define Bb 2

// ---------------- WMMA fragment helpers (ISA 7.12.2 layouts) ----------------

union HFrag { v16h v; v8h p[2]; };

// A: 16x32 f16 tile from LDS, row-major, row stride `astride` halfs (x8 bytes %16==0).
// Lane (m=lane&15, g=lane>>4): elements 0..7 = K[kc*32+8g .. +8),
// elements 8..15 = K[kc*32+16+8g .. +8)  -> two contiguous b128 loads.
__device__ inline v16h load_a_frag(const _Float16* tile, int kc, int astride, int lane) {
  int m = lane & 15, g = lane >> 4;
  const _Float16* base = tile + m * astride + kc * 32 + 8 * g;
  HFrag f;
  f.p[0] = *(const v8h*)(base);
  f.p[1] = *(const v8h*)(base + 16);
  return f.v;
}

// B: 32x16 f16 tile from LDS weight stored TRANSPOSED [N][K], row stride `kstride`.
// Lane (n=(lane&15)+16*nt, g=lane>>4): elements 0..7 = K[kc*32+8g .. +8),
// elements 8..15 = K[kc*32+16+8g .. +8)  -> two b128 loads.
__device__ inline v16h load_b_frag(const _Float16* wT, int kc, int nt, int kstride, int lane) {
  int n = (lane & 15) + 16 * nt, g = lane >> 4;
  const _Float16* base = wT + n * kstride + kc * 32 + 8 * g;
  HFrag f;
  f.p[0] = *(const v8h*)(base);
  f.p[1] = *(const v8h*)(base + 16);
  return f.v;
}

__device__ inline float silu(float x) { return x / (1.0f + __expf(-x)); }

// out[16][96] = silu(A[16xK] * W^T[96xK]^T + bias), f16 out in LDS
template <int KCHUNKS>
__device__ inline void gemm_tile_silu(const _Float16* A, int astride,
                                      const _Float16* WT, int kstride,
                                      const float* bias, _Float16* out, int lane) {
  asm volatile("s_wait_dscnt 0x0" ::: "memory");
#pragma unroll
  for (int nt = 0; nt < 6; ++nt) {
    v8f c = {};
#pragma unroll
    for (int kc = 0; kc < KCHUNKS; ++kc) {
      v16h a = load_a_frag(A, kc, astride, lane);
      v16h b = load_b_frag(WT, kc, nt, kstride, lane);
      c = __builtin_amdgcn_wmma_f32_16x16x32_f16(false, a, false, b, (short)0, c,
                                                 false, false);
    }
    int n = (lane & 15) + 16 * nt, g = lane >> 4;
#pragma unroll
    for (int r = 0; r < 8; ++r) {
      float xv = c[r] + bias[n];
      out[(r + 8 * g) * Hdim + n] = (_Float16)silu(xv);
    }
  }
  asm volatile("s_wait_dscnt 0x0" ::: "memory");
}

// h[row0+m][n] += A[16x96] * W + bias (node MLP residual, no silu)
__device__ inline void gemm_tile_addh(const _Float16* A, const _Float16* WT,
                                      const float* bias, float* h, int row0,
                                      int lane) {
  asm volatile("s_wait_dscnt 0x0" ::: "memory");
#pragma unroll
  for (int nt = 0; nt < 6; ++nt) {
    v8f c = {};
#pragma unroll
    for (int kc = 0; kc < 3; ++kc) {
      v16h a = load_a_frag(A, kc, Hdim, lane);
      v16h b = load_b_frag(WT, kc, nt, Hdim, lane);
      c = __builtin_amdgcn_wmma_f32_16x16x32_f16(false, a, false, b, (short)0, c,
                                                 false, false);
    }
    int n = (lane & 15) + 16 * nt, g = lane >> 4;
#pragma unroll
    for (int r = 0; r < 8; ++r) {
      size_t off = (size_t)(row0 + r + 8 * g) * Hdim + n;
      h[off] += c[r] + bias[n];
    }
  }
}

// ---------------- Setup kernels ----------------

__global__ void feats_kernel(const int* __restrict__ tgroups,
                             const int* __restrict__ tindices,
                             const int* __restrict__ ntypes,
                             const float* __restrict__ seqpos,
                             const float* __restrict__ protein,
                             const float* __restrict__ nucleo,
                             const float* __restrict__ ligand,
                             const float* __restrict__ modality,
                             const float* __restrict__ Wp0, const float* __restrict__ bp0,
                             const float* __restrict__ Wp1, const float* __restrict__ bp1,
                             float* __restrict__ h) {
  __shared__ float hid[Hdim];
  int n = blockIdx.x, t = threadIdx.x;
  float p = seqpos[n];
  hid[t] = silu(Wp0[t] * p + bp0[t]);
  __syncthreads();
  float acc = bp1[t];
  for (int j = 0; j < Hdim; ++j) acc += hid[j] * Wp1[j * Hdim + t];
  int g = tgroups[n], idx = tindices[n];
  float e;
  if (g == 0)      e = protein[idx * Hdim + t];
  else if (g == 1) e = nucleo[idx * Hdim + t];
  else if (g == 2) e = ligand[idx * Hdim + t];
  else             e = 0.0f;
  float v = e + modality[ntypes[n] * Hdim + t] + acc;
  h[(size_t)n * Hdim + t] = v;
  h[(size_t)(Nn + n) * Hdim + t] = v;
}

__global__ void temb_kernel(const float* __restrict__ sigma,
                            const float* __restrict__ Wt0, const float* __restrict__ bt0,
                            const float* __restrict__ Wt1, const float* __restrict__ bt1,
                            float* __restrict__ temb) {
  __shared__ float emb[Tdim];
  __shared__ float hid[Hdim];
  int b = blockIdx.x, t = threadIdx.x;
  float s = sigma[b];
  if (t < Tdim) {
    int i = t & 15;
    float fr = __expf((float)i * (6.9077552789821368f / 15.0f)); // ln(1000)/15
    float a = s * fr;
    emb[t] = (t < 16) ? __sinf(a) : __cosf(a);
  }
  __syncthreads();
  float acc = bt0[t];
  for (int k = 0; k < Tdim; ++k) acc += emb[k] * Wt0[k * Hdim + t];
  hid[t] = silu(acc);
  __syncthreads();
  if (t < Tdim) {
    float o = bt1[t];
    for (int j = 0; j < Hdim; ++j) o += hid[j] * Wt1[j * Tdim + t];
    temb[b * Tdim + t] = o;
  }
}

// ---------------- Edge kernel ----------------

#define EK_WAVES 4
#define EK_KPAD 352
#define EK_W_HALF (EK_KPAD * Hdim + Hdim * Hdim + Hdim * Hdim) // 52224 halfs
#define EK_F (Hdim * 4 + Tdim)                                 // 416 floats
#define EK_WAVE_HALF (16 * EK_KPAD + 16 * Hdim)                // 7168 halfs
#define EK_WAVE_F 128
#define EK_SMEM ((size_t)EK_W_HALF * 2 + (size_t)EK_WAVES * EK_WAVE_HALF * 2 + \
                 (size_t)EK_F * 4 + (size_t)EK_WAVES * EK_WAVE_F * 4)

__global__ __launch_bounds__(EK_WAVES * 32) void edge_kernel(
    const float* __restrict__ x, const float* __restrict__ h,
    const int* __restrict__ eidx, const int* __restrict__ etyp,
    const float* __restrict__ temb, const float* __restrict__ eemb,
    const float* __restrict__ We0, const float* __restrict__ be0,
    const float* __restrict__ We1, const float* __restrict__ be1,
    const float* __restrict__ Wc0, const float* __restrict__ bc0,
    const float* __restrict__ Wc1, const float* __restrict__ bc1,
    float* __restrict__ coord_agg, float* __restrict__ feat_agg) {
  extern __shared__ char smem[];
  int tid = threadIdx.x, lane = tid & 31, wave = tid >> 5;
  int b = blockIdx.y;

  _Float16* sW0 = (_Float16*)smem;                 // transposed [96][352]
  _Float16* sW1 = sW0 + EK_KPAD * Hdim;            // transposed [96][96]
  _Float16* sWc = sW1 + Hdim * Hdim;               // transposed [96][96]
  _Float16* waveH = sWc + Hdim * Hdim + wave * EK_WAVE_HALF;
  _Float16* bufA = waveH;                          // 16x352 (then reused 16x96)
  _Float16* bufM = waveH + 16 * EK_KPAD;           // 16x96
  float* fbase = (float*)(sW0 + EK_W_HALF + EK_WAVES * EK_WAVE_HALF);
  float* sbe0 = fbase;
  float* sbe1 = fbase + Hdim;
  float* sbc0 = fbase + 2 * Hdim;
  float* sWc1 = fbase + 3 * Hdim;
  float* stmb = fbase + 4 * Hdim;                  // 32
  float* wf = fbase + EK_F + wave * EK_WAVE_F;
  int* ssrc = (int*)wf;
  int* sdst = ssrc + 16;
  int* setp = sdst + 16;
  float* srel = (float*)(setp + 16);               // 48
  float* sdist = srel + 48;                        // 16
  float* sscal = sdist + 16;                       // 16

  // stage weights transposed to [N][K] for contiguous B-fragments
  for (int i = tid; i < Hdim * EK_KPAD; i += blockDim.x) {
    int n = i / EK_KPAD, k = i % EK_KPAD;
    sW0[i] = (_Float16)((k < 336) ? We0[k * Hdim + n] : 0.0f);
  }
  for (int i = tid; i < Hdim * Hdim; i += blockDim.x) {
    int n = i / Hdim, k = i % Hdim;
    sW1[i] = (_Float16)We1[k * Hdim + n];
    sWc[i] = (_Float16)Wc0[k * Hdim + n];
  }
  for (int i = tid; i < Hdim; i += blockDim.x) {
    sbe0[i] = be0[i]; sbe1[i] = be1[i]; sbc0[i] = bc0[i]; sWc1[i] = Wc1[i];
  }
  if (tid < Tdim) stmb[tid] = temb[b * Tdim + tid];
  __syncthreads();

  const float bc1v = bc1[0];
  const size_t xb = (size_t)b * Nn * 3;
  const size_t hb = (size_t)b * Nn * Hdim;

  int tiles = Ee / 16;
  for (int tile = blockIdx.x * EK_WAVES + wave; tile < tiles;
       tile += gridDim.x * EK_WAVES) {
    int e0 = tile * 16;
    if (lane < 16) {
      int e = e0 + lane;
      int es = eidx[2 * e], ed = eidx[2 * e + 1];
      ssrc[lane] = es; sdst[lane] = ed; setp[lane] = etyp[e];
      float dx = x[xb + es * 3 + 0] - x[xb + ed * 3 + 0];
      float dy = x[xb + es * 3 + 1] - x[xb + ed * 3 + 1];
      float dz = x[xb + es * 3 + 2] - x[xb + ed * 3 + 2];
      srel[lane * 3 + 0] = dx; srel[lane * 3 + 1] = dy; srel[lane * 3 + 2] = dz;
      sdist[lane] = fmaxf(sqrtf(dx * dx + dy * dy + dz * dz), 1e-6f);
    }
    asm volatile("s_wait_dscnt 0x0" ::: "memory");

    // stage msg_in: [h_src 96 | h_dst 96 | edge_emb 96 | t 32 | rbf 16 | pad 16]
    for (int idx = lane; idx < 16 * EK_KPAD; idx += 32) {
      int m = idx / EK_KPAD, f = idx % EK_KPAD;
      float v;
      if (f < 96)        v = h[hb + (size_t)ssrc[m] * Hdim + f];
      else if (f < 192)  v = h[hb + (size_t)sdst[m] * Hdim + (f - 96)];
      else if (f < 288)  v = eemb[setp[m] * Hdim + (f - 192)];
      else if (f < 320)  v = stmb[f - 288];
      else if (f < 336) {
        float d = sdist[m] - (12.0f / 15.0f) * (float)(f - 320);
        v = __expf(-0.5f * d * d);
      } else v = 0.0f;
      bufA[m * EK_KPAD + f] = (_Float16)v;
    }

    gemm_tile_silu<11>(bufA, EK_KPAD, sW0, EK_KPAD, sbe0, bufM, lane); // 336->96
    gemm_tile_silu<3>(bufM, Hdim, sW1, Hdim, sbe1, bufA, lane);        // messages
    gemm_tile_silu<3>(bufA, Hdim, sWc, Hdim, sbc0, bufM, lane);        // coord hid

    if (lane < 16) {
      float s = bc1v;
      for (int j = 0; j < Hdim; ++j) s += (float)bufM[lane * Hdim + j] * sWc1[j];
      sscal[lane] = s / (sdist[lane] + 1.0f);
    }
    asm volatile("s_wait_dscnt 0x0" ::: "memory");

    for (int idx = lane; idx < 48; idx += 32) {
      int m = idx / 3, d = idx % 3;
      atomicAdd(&coord_agg[xb + (size_t)ssrc[m] * 3 + d], sscal[m] * srel[idx]);
    }
    for (int idx = lane; idx < 16 * Hdim; idx += 32) {
      int m = idx / Hdim, c = idx % Hdim;
      atomicAdd(&feat_agg[hb + (size_t)ssrc[m] * Hdim + c],
                (float)bufA[m * Hdim + c]);
    }
    asm volatile("s_wait_dscnt 0x0" ::: "memory");
  }
}

// ---------------- Node kernel ----------------

#define NK_WAVES 4
#define NK_W_HALF (192 * Hdim + Hdim * Hdim)
#define NK_WAVE_HALF (16 * 192 + 16 * Hdim)
#define NK_SMEM ((size_t)NK_W_HALF * 2 + (size_t)NK_WAVES * NK_WAVE_HALF * 2 + \
                 2 * Hdim * 4)

__global__ __launch_bounds__(NK_WAVES * 32) void node_kernel(
    float* __restrict__ x, float* __restrict__ h,
    const float* __restrict__ coord_agg, const float* __restrict__ feat_agg,
    const float* __restrict__ Wn0, const float* __restrict__ bn0,
    const float* __restrict__ Wn1, const float* __restrict__ bn1) {
  extern __shared__ char smem[];
  int tid = threadIdx.x, lane = tid & 31, wave = tid >> 5;
  _Float16* sW0 = (_Float16*)smem;                  // transposed [96][192]
  _Float16* sW1 = sW0 + 192 * Hdim;                 // transposed [96][96]
  _Float16* waveH = sW1 + Hdim * Hdim + wave * NK_WAVE_HALF;
  _Float16* bufA = waveH;                            // 16x192
  _Float16* bufM = waveH + 16 * 192;                 // 16x96
  float* sb0 = (float*)(sW0 + NK_W_HALF + NK_WAVES * NK_WAVE_HALF);
  float* sb1 = sb0 + Hdim;

  for (int i = tid; i < Hdim * 192; i += blockDim.x) {
    int n = i / 192, k = i % 192;
    sW0[i] = (_Float16)Wn0[k * Hdim + n];
  }
  for (int i = tid; i < Hdim * Hdim; i += blockDim.x) {
    int n = i / Hdim, k = i % Hdim;
    sW1[i] = (_Float16)Wn1[k * Hdim + n];
  }
  for (int i = tid; i < Hdim; i += blockDim.x) { sb0[i] = bn0[i]; sb1[i] = bn1[i]; }
  __syncthreads();

  int tiles = (Bb * Nn) / 16;
  for (int tile = blockIdx.x * NK_WAVES + wave; tile < tiles;
       tile += gridDim.x * NK_WAVES) {
    int r0 = tile * 16;
    for (int idx = lane; idx < 16 * 192; idx += 32) {
      int m = idx / 192, f = idx % 192;
      size_t row = (size_t)(r0 + m);
      float v = (f < 96) ? h[row * Hdim + f] : feat_agg[row * Hdim + (f - 96)];
      bufA[m * 192 + f] = (_Float16)v;
    }
    gemm_tile_silu<6>(bufA, 192, sW0, 192, sb0, bufM, lane);
    gemm_tile_addh(bufM, sW1, sb1, h, r0, lane);
    for (int idx = lane; idx < 48; idx += 32) {
      size_t off = (size_t)r0 * 3 + idx;
      x[off] += coord_agg[off];
    }
  }
}

// ---------------- Output kernel ----------------

#define OK_WAVES 4
#define OK_WAVE_HALF (16 * Hdim * 2)
#define OK_SMEM ((size_t)Hdim * Hdim * 2 + (size_t)OK_WAVES * OK_WAVE_HALF * 2 + \
                 (Hdim + Hdim * 3 + 4) * 4)

__global__ __launch_bounds__(OK_WAVES * 32) void out_kernel(
    const float* __restrict__ x, const float* __restrict__ h,
    const float* __restrict__ Wo0, const float* __restrict__ bo0,
    const float* __restrict__ Wo1, const float* __restrict__ bo1,
    float* __restrict__ out) {
  extern __shared__ char smem[];
  int tid = threadIdx.x, lane = tid & 31, wave = tid >> 5;
  _Float16* sW0 = (_Float16*)smem;                    // transposed [96][96]
  _Float16* waveH = sW0 + Hdim * Hdim + wave * OK_WAVE_HALF;
  _Float16* bufA = waveH;
  _Float16* bufM = waveH + 16 * Hdim;
  float* sb0 = (float*)(sW0 + Hdim * Hdim + OK_WAVES * OK_WAVE_HALF);
  float* sW1 = sb0 + Hdim;                            // 96*3
  float* sb1 = sW1 + Hdim * 3;                        // 3

  for (int i = tid; i < Hdim * Hdim; i += blockDim.x) {
    int n = i / Hdim, k = i % Hdim;
    sW0[i] = (_Float16)Wo0[k * Hdim + n];
  }
  for (int i = tid; i < Hdim; i += blockDim.x) sb0[i] = bo0[i];
  for (int i = tid; i < Hdim * 3; i += blockDim.x) sW1[i] = Wo1[i];
  if (tid < 3) sb1[tid] = bo1[tid];
  __syncthreads();

  int tiles = (Bb * Nn) / 16;
  for (int tile = blockIdx.x * OK_WAVES + wave; tile < tiles;
       tile += gridDim.x * OK_WAVES) {
    int r0 = tile * 16;
    for (int idx = lane; idx < 16 * Hdim; idx += 32) {
      int m = idx / Hdim, f = idx % Hdim;
      bufA[idx] = (_Float16)h[(size_t)(r0 + m) * Hdim + f];
    }
    gemm_tile_silu<3>(bufA, Hdim, sW0, Hdim, sb0, bufM, lane);
    for (int idx = lane; idx < 48; idx += 32) {
      int m = idx / 3, c = idx % 3;
      float d = sb1[c];
      for (int j = 0; j < Hdim; ++j) d += (float)bufM[m * Hdim + j] * sW1[j * 3 + c];
      size_t off = (size_t)(r0 + m) * 3 + c;
      out[off] = x[off] + d;
    }
    asm volatile("s_wait_dscnt 0x0" ::: "memory");
  }
}

// ---------------- Launch ----------------

extern "C" void kernel_launch(void* const* d_in, const int* in_sizes, int n_in,
                              void* d_out, int out_size, void* d_ws, size_t ws_size,
                              hipStream_t stream) {
  const float* noisy   = (const float*)d_in[0];
  const float* sigma   = (const float*)d_in[1];
  const int*   tgroups = (const int*)d_in[2];
  const int*   tindices= (const int*)d_in[3];
  const int*   ntypes  = (const int*)d_in[4];
  const float* seqpos  = (const float*)d_in[5];
  const int*   eindex  = (const int*)d_in[6];
  const int*   etypes  = (const int*)d_in[7];
  auto P = [&](int i) { return (const float*)d_in[i]; };
  // params pytree (sorted dict keys): layers[4] x {coord_mlp(w0,b0,w1,b1),
  // edge_emb, edge_mlp(w0,b0,w1,b1), node_mlp(w0,b0,w1,b1)} then ligand,
  // modality, nucleotide, out_mlp, position_mlp, protein, time_mlp.
  const float* ligand = P(60); const float* modality = P(61); const float* nucleo = P(62);
  const float *Wo0 = P(63), *bo0 = P(64), *Wo1 = P(65), *bo1 = P(66);
  const float *Wp0 = P(67), *bp0 = P(68), *Wp1 = P(69), *bp1 = P(70);
  const float* protein = P(71);
  const float *Wt0 = P(72), *bt0 = P(73), *Wt1 = P(74), *bt1 = P(75);

  float* ws = (float*)d_ws;
  float* x         = ws;             ws += (size_t)Bb * Nn * 3;
  float* h         = ws;             ws += (size_t)Bb * Nn * Hdim;
  float* coord_agg = ws;             ws += (size_t)Bb * Nn * 3;
  float* feat_agg  = ws;             ws += (size_t)Bb * Nn * Hdim;
  float* temb      = ws;             ws += (size_t)Bb * Tdim;

  hipMemcpyAsync(x, noisy, (size_t)Bb * Nn * 3 * sizeof(float),
                 hipMemcpyDeviceToDevice, stream);
  feats_kernel<<<Nn, Hdim, 0, stream>>>(tgroups, tindices, ntypes, seqpos,
                                        protein, nucleo, ligand, modality,
                                        Wp0, bp0, Wp1, bp1, h);
  temb_kernel<<<Bb, Hdim, 0, stream>>>(sigma, Wt0, bt0, Wt1, bt1, temb);

  for (int l = 0; l < NLAYERS; ++l) {
    int base = 8 + l * 13;
    const float *Wc0 = P(base + 0), *bc0 = P(base + 1);
    const float *Wc1 = P(base + 2), *bc1 = P(base + 3);
    const float *eemb = P(base + 4);
    const float *We0 = P(base + 5), *be0 = P(base + 6);
    const float *We1 = P(base + 7), *be1 = P(base + 8);
    const float *Wn0 = P(base + 9), *bn0 = P(base + 10);
    const float *Wn1 = P(base + 11), *bn1 = P(base + 12);

    hipMemsetAsync(coord_agg, 0, (size_t)Bb * Nn * 3 * sizeof(float), stream);
    hipMemsetAsync(feat_agg, 0, (size_t)Bb * Nn * Hdim * sizeof(float), stream);

    edge_kernel<<<dim3(256, Bb), EK_WAVES * 32, EK_SMEM, stream>>>(
        x, h, eindex, etypes, temb, eemb, We0, be0, We1, be1, Wc0, bc0, Wc1,
        bc1, coord_agg, feat_agg);
    node_kernel<<<dim3(128, 1), NK_WAVES * 32, NK_SMEM, stream>>>(
        x, h, coord_agg, feat_agg, Wn0, bn0, Wn1, bn1);
  }

  out_kernel<<<dim3(128, 1), OK_WAVES * 32, OK_SMEM, stream>>>(
      x, h, Wo0, bo0, Wo1, bo1, (float*)d_out);
}
